// ADC_10033043604239
// MI455X (gfx1250) — matrix operand order repaired and checked by reading, not compiled, not run on hardware
//
#include <hip/hip_runtime.h>

// ---------------------------------------------------------------------------
// MI455X (gfx1250) fused dynamic-conv block, round 2.
// Compute-bound (~29 GFLOP vs ~67MB HBM) -> bf16 WMMA (16x16x32, f32 acc) for
// conv0 / dynamic-filter GEMM / convf. Round-2 change: 2-wide pixel blocking
// so each A (weight) fragment feeds 2 WMMAs -> ~1.5 b128 loads per WMMA
// (was 2.5), cutting the s_wait_loadcnt stalls seen in round-1 disasm.
// ---------------------------------------------------------------------------

typedef __attribute__((ext_vector_type(16))) __bf16       v16bf;
typedef __attribute__((ext_vector_type(8)))  float        v8f;
typedef __attribute__((ext_vector_type(4)))  unsigned int v4u;

#define DEV static __device__ __forceinline__

constexpr int Bb = 8, Cc = 64, Hh = 128, Ww = 128;
constexpr int HP = Hh + 2, WP = Ww + 2;           // padded spatial dims
constexpr int NPAD   = Bb * HP * WP * Cc;         // 8,652,800 bf16 elems
constexpr int NDENSE = Bb * Hh * Ww * Cc;         // 8,388,608

// A fragment, 16x32 bf16 (M=cout rows, K=cin). Caller applies +8*(lane>=16).
// lanes 0-15: V0-3 = K 0..7, V4-7 = K 16..23 ; lanes 16-31: +8.
DEV v16bf load_fragA(const __bf16* p) {
  union { v16bf f; v4u u[2]; } r;
  r.u[0] = *(const v4u*)(p);
  r.u[1] = *(const v4u*)(p + 16);
  return r.f;
}
// B fragment, 32x16 bf16 (K=cin rows, N=pixel cols). Caller applies +8*(lane>=16).
// lanes 0-15: V0-3 = K 0..7, V4-7 = K 16..23 ; lanes 16-31: +8.
DEV v16bf load_fragB(const __bf16* p) {
  union { v16bf f; v4u u[2]; } r;
  r.u[0] = *(const v4u*)(p);
  r.u[1] = *(const v4u*)(p + 16);
  return r.f;
}
DEV v8f wmma_bf16(v16bf a, v16bf b, v8f c) {
  return __builtin_amdgcn_wmma_f32_16x16x32_bf16(
      /*neg_a=*/false, a, /*neg_b=*/false, b,
      /*c_mod=*/(short)0, c, /*reuse_a=*/false, /*reuse_b=*/false);
}

// ---------------------------------------------------------------------------
// Staging kernels
// ---------------------------------------------------------------------------

// x (fp32 NCHW) -> xpad (bf16 NHWC, zero halo)
__global__ void stage_x_kernel(const float* __restrict__ x, __bf16* __restrict__ xpad) {
  int t = blockIdx.x * 256 + threadIdx.x;
  if (t >= NPAD) return;
  int c  = t & 63;
  int wp = (t >> 6) % WP;
  int hp = (t / (64 * WP)) % HP;
  int b  = t / (64 * WP * HP);
  __bf16 v = (__bf16)0.0f;
  if (hp >= 1 && hp <= Hh && wp >= 1 && wp <= Ww)
    v = (__bf16)x[(((size_t)(b * Cc + c) * Hh + (hp - 1)) << 7) + (wp - 1)];
  xpad[t] = v;
}

__global__ void zero_bf16_kernel(__bf16* __restrict__ p, int n) {
  int t = blockIdx.x * 256 + threadIdx.x;
  if (t < n) p[t] = (__bf16)0.0f;
}

// w0 / wf (Cout,Cin,3,3) -> [pos][cout][cin] bf16 ; w3 (576,64,1,1) -> [o][cin] bf16
__global__ void stage_w_kernel(const float* __restrict__ w0, const float* __restrict__ wf,
                               const float* __restrict__ w3,
                               __bf16* __restrict__ w0s, __bf16* __restrict__ wfs,
                               __bf16* __restrict__ w3s) {
  int t = blockIdx.x * 256 + threadIdx.x;
  if (t >= 3 * 36864) return;
  if (t < 36864) {
    int d = t, ci = d & 63, co = (d >> 6) & 63, p = d >> 12;
    w0s[d] = (__bf16)w0[(co * 64 + ci) * 9 + p];
  } else if (t < 2 * 36864) {
    int d = t - 36864, ci = d & 63, co = (d >> 6) & 63, p = d >> 12;
    wfs[d] = (__bf16)wf[(co * 64 + ci) * 9 + p];
  } else {
    int d = t - 2 * 36864;
    w3s[d] = (__bf16)w3[d];
  }
}

// ---------------------------------------------------------------------------
// Dense 3x3 conv, 64->64, implicit GEMM on WMMA.
// One wave: 32 contiguous pixels (2 N-tiles) x all 64 cout -> 8 accumulators.
// A fragments reused across both pixel tiles (halves weight-load traffic).
// MODE 0: conv0 -> bf16 padded NHWC (x0pad), bias folded in.
// MODE 1: convf -> fp32 NCHW d_out = acc + bias + residual x.
// ---------------------------------------------------------------------------
template <int MODE>
__global__ __launch_bounds__(128)
void conv3x3_wmma_kernel(const __bf16* __restrict__ inpad,
                         const __bf16* __restrict__ wst,   // [9][64][64] bf16
                         const float* __restrict__ bias,
                         __bf16* __restrict__ outpad,      // MODE 0
                         float* __restrict__ outf,         // MODE 1
                         const float* __restrict__ xres) { // MODE 1
  const int lane = threadIdx.x & 31;
  const int wid  = blockIdx.x * 4 + (threadIdx.x >> 5);  // 4096 waves total
  const int w0p  = (wid & 3) * 32;                       // 32-pixel strip
  const int h    = (wid >> 2) & 127;
  const int b    = wid >> 9;
  const int n    = lane & 15;
  const int hf   = lane >> 4;
  const int mofs = hf * 8;

  v8f acc[4][2];
#pragma unroll
  for (int t = 0; t < 4; ++t)
#pragma unroll
    for (int r = 0; r < 8; ++r) {
      float bv = bias[t * 16 + mofs + r];
      acc[t][0][r] = bv;
      acc[t][1][r] = bv;
    }

#pragma unroll
  for (int p = 0; p < 9; ++p) {
    const int i = p / 3, j = p % 3;
    const __bf16* brow =
        inpad + (((size_t)((b * HP + h + i) * WP + (w0p + j + n))) << 6) + hf * 8;
    if (p < 8) {  // prefetch next tap row into L0 (global_prefetch_b8)
      const int i2 = (p + 1) / 3, j2 = (p + 1) % 3;
      __builtin_prefetch(
          inpad + (((size_t)((b * HP + h + i2) * WP + (w0p + j2 + n))) << 6), 0, 0);
    }
#pragma unroll
    for (int cb = 0; cb < 64; cb += 32) {
      v16bf Bf0 = load_fragB(brow + cb);
      v16bf Bf1 = load_fragB(brow + (16 << 6) + cb);  // +16 pixels
#pragma unroll
      for (int t = 0; t < 4; ++t) {
        v16bf Af = load_fragA(wst + (((size_t)(p * 64 + t * 16 + n)) << 6) + cb + hf * 8);
        acc[t][0] = wmma_bf16(Af, Bf0, acc[t][0]);
        acc[t][1] = wmma_bf16(Af, Bf1, acc[t][1]);
      }
    }
  }

  if (MODE == 0) {
#pragma unroll
    for (int pt = 0; pt < 2; ++pt) {
      size_t pofs = ((size_t)((b * HP + h + 1) * WP + (w0p + pt * 16 + n + 1))) << 6;
#pragma unroll
      for (int t = 0; t < 4; ++t) {
        union { v4u u; __bf16 h8[8]; } pk;
#pragma unroll
        for (int r = 0; r < 8; ++r) pk.h8[r] = (__bf16)acc[t][pt][r];
        *(v4u*)(outpad + pofs + t * 16 + mofs) = pk.u;
      }
    }
  } else {
#pragma unroll
    for (int pt = 0; pt < 2; ++pt)
#pragma unroll
      for (int t = 0; t < 4; ++t)
#pragma unroll
        for (int r = 0; r < 8; ++r) {
          int c = t * 16 + mofs + r;
          size_t o = (((size_t)(b * Cc + c) * Hh + h) << 7) + (w0p + pt * 16 + n);
          outf[o] = acc[t][pt][r] + xres[o];
        }
  }
}

// ---------------------------------------------------------------------------
// mean over H,W of x0 -> m[b][c]
// ---------------------------------------------------------------------------
__global__ void mean_kernel(const __bf16* __restrict__ x0pad, float* __restrict__ m) {
  __shared__ float sm[256];
  int bc = blockIdx.x, b = bc >> 6, c = bc & 63;
  float s = 0.0f;
  for (int p = threadIdx.x; p < Hh * Ww; p += 256) {
    int h = p >> 7, w = p & 127;
    s += (float)x0pad[(((size_t)((b * HP + h + 1) * WP + (w + 1))) << 6) + c];
  }
  sm[threadIdx.x] = s;
  __syncthreads();
  for (int st = 128; st > 0; st >>= 1) {
    if ((int)threadIdx.x < st) sm[threadIdx.x] += sm[threadIdx.x + st];
    __syncthreads();
  }
  if (threadIdx.x == 0) m[bc] = sm[0] * (1.0f / (Hh * Ww));
}

// ECA: 1D conv over the channel axis (K=3, zero pad) -> a[b][c]
__global__ void eca_kernel(const float* __restrict__ m, const float* __restrict__ w2,
                           const float* __restrict__ b2, float* __restrict__ a) {
  int id = blockIdx.x * 256 + threadIdx.x;
  if (id >= Bb * Cc) return;
  int b = id >> 6, c = id & 63;
  float s = b2[0];
#pragma unroll
  for (int d = 0; d < 3; ++d) {
    int cc = c + d - 1;
    if (cc >= 0 && cc < Cc) s += w2[d] * m[b * Cc + cc];
  }
  a[id] = s;
}

// depthwise 3x3 on x0 (+b1) + broadcast channel attention -> dyn (bf16 NHWC)
__global__ void dw_kernel(const __bf16* __restrict__ x0pad, const float* __restrict__ w1,
                          const float* __restrict__ b1, const float* __restrict__ av,
                          __bf16* __restrict__ dyn) {
  int t = blockIdx.x * 256 + threadIdx.x;
  if (t >= NDENSE) return;
  int c = t & 63, w = (t >> 6) & 127, h = (t >> 13) & 127, b = t >> 20;
  float s = b1[c] + av[(b << 6) | c];
#pragma unroll
  for (int k = 0; k < 9; ++k) {
    int i = k / 3, j = k % 3;
    s += w1[c * 9 + k] *
         (float)x0pad[(((size_t)((b * HP + h + i) * WP + (w + j))) << 6) + c];
  }
  dyn[t] = (__bf16)s;  // NHWC index == t
}

// ---------------------------------------------------------------------------
// 1x1 (576x64) GEMM -> per-pixel dynamic 3x3 conv over unfold(x0) -> LeakyReLU
// One wave: 32 pixels (2 N-tiles). fil computed in 4 groups of 144 rows
// (16ch x 9 taps), spilled through per-wave LDS (18KB), contracted in place.
// 2 waves / block -> 36KB static LDS.
// ---------------------------------------------------------------------------
__global__ __launch_bounds__(64)
void dynconv_kernel(const __bf16* __restrict__ dyn, const __bf16* __restrict__ w3s,
                    const float* __restrict__ b3, const __bf16* __restrict__ x0pad,
                    __bf16* __restrict__ tpad) {
  __shared__ float lds[2 * 144 * 32];  // 36 KB: one 144x32 tile per wave
  const int lane = threadIdx.x & 31;
  const int wid  = blockIdx.x * 2 + (threadIdx.x >> 5);  // 4096 waves
  const int w0p  = (wid & 3) * 32;
  const int h    = (wid >> 2) & 127;
  const int b    = wid >> 9;
  const int n    = lane & 15;
  const int hf   = lane >> 4;
  const int mofs = hf * 8;
  float* L = lds + (threadIdx.x >> 5) * (144 * 32);

  // B fragments (dyn activations, K=cin) for both pixel tiles; reused by all
  // 4 channel groups and all 9 row-tiles.
  const __bf16* dbase =
      dyn + (((size_t)((b * Hh + h) * Ww + (w0p + n))) << 6) + hf * 8;
  v16bf B0a = load_fragB(dbase + 0);
  v16bf B1a = load_fragB(dbase + 32);
  v16bf B0b = load_fragB(dbase + (16 << 6) + 0);
  v16bf B1b = load_fragB(dbase + (16 << 6) + 32);

  for (int g = 0; g < 4; ++g) {
    // fil rows [g*144, g*144+144) for these 32 pixels
#pragma unroll
    for (int t9 = 0; t9 < 9; ++t9) {
      int rowb = g * 144 + t9 * 16;
      v8f acc0, acc1;
#pragma unroll
      for (int r = 0; r < 8; ++r) {
        float bv = b3[rowb + mofs + r];
        acc0[r] = bv;
        acc1[r] = bv;
      }
      const __bf16* a0 = w3s + (((size_t)(rowb + n)) << 6) + hf * 8;
      v16bf Af0 = load_fragA(a0);
      v16bf Af1 = load_fragA(a0 + 32);
      acc0 = wmma_bf16(Af0, B0a, acc0);
      acc0 = wmma_bf16(Af1, B1a, acc0);
      acc1 = wmma_bf16(Af0, B0b, acc1);
      acc1 = wmma_bf16(Af1, B1b, acc1);
#pragma unroll
      for (int r = 0; r < 8; ++r) {
        L[(t9 * 16 + mofs + r) * 32 + n] = acc0[r];
        L[(t9 * 16 + mofs + r) * 32 + 16 + n] = acc1[r];
      }
    }
    // contract with unfold(x0): channel c = g*16 + (mofs + r), pixel pt*16+n
#pragma unroll
    for (int pt = 0; pt < 2; ++pt) {
      union { v4u u; __bf16 h8[8]; } pk;
#pragma unroll
      for (int r = 0; r < 8; ++r) {
        int cl = mofs + r;
        float s = 0.0f;
#pragma unroll
        for (int k = 0; k < 9; ++k) {
          int i = k / 3, j = k % 3;
          s += L[(cl * 9 + k) * 32 + pt * 16 + n] *
               (float)x0pad[(((size_t)((b * HP + h + i) * WP +
                                       (w0p + pt * 16 + n + j))) << 6) +
                            g * 16 + cl];
        }
        s = (s >= 0.0f) ? s : 0.2f * s;  // LeakyReLU(0.2)
        pk.h8[r] = (__bf16)s;
      }
      *(v4u*)(tpad + (((size_t)((b * HP + h + 1) * WP + (w0p + pt * 16 + n + 1))) << 6) +
              g * 16 + mofs) = pk.u;
    }
  }
}

// ---------------------------------------------------------------------------
// Launcher
// ---------------------------------------------------------------------------
extern "C" void kernel_launch(void* const* d_in, const int* in_sizes, int n_in,
                              void* d_out, int out_size, void* d_ws, size_t ws_size,
                              hipStream_t stream) {
  const float* x  = (const float*)d_in[0];
  const float* w0 = (const float*)d_in[1];
  const float* b0 = (const float*)d_in[2];
  const float* w1 = (const float*)d_in[3];
  const float* b1 = (const float*)d_in[4];
  const float* w2 = (const float*)d_in[5];
  const float* b2 = (const float*)d_in[6];
  const float* w3 = (const float*)d_in[7];
  const float* b3 = (const float*)d_in[8];
  const float* wf = (const float*)d_in[9];
  const float* bf = (const float*)d_in[10];
  float* out = (float*)d_out;

  // Workspace layout (all offsets 256B aligned; ~66 MB total, L2-resident)
  char* ws = (char*)d_ws;
  size_t o = 0;
  __bf16* xpad  = (__bf16*)(ws + o); o += (size_t)NPAD * 2;      // 17,305,600
  __bf16* x0pad = (__bf16*)(ws + o); o += (size_t)NPAD * 2;
  __bf16* tpad  = (__bf16*)(ws + o); o += (size_t)NPAD * 2;
  __bf16* dyn   = (__bf16*)(ws + o); o += (size_t)NDENSE * 2;    // 16,777,216
  __bf16* w0s   = (__bf16*)(ws + o); o += 36864 * 2;
  __bf16* wfs   = (__bf16*)(ws + o); o += 36864 * 2;
  __bf16* w3s   = (__bf16*)(ws + o); o += 36864 * 2;
  float*  mbuf  = (float*)(ws + o);  o += Bb * Cc * 4;
  float*  abuf  = (float*)(ws + o);  o += Bb * Cc * 4;

  const int padBlocks = (NPAD + 255) / 256;      // 33800
  stage_x_kernel<<<padBlocks, 256, 0, stream>>>(x, xpad);
  zero_bf16_kernel<<<padBlocks, 256, 0, stream>>>(x0pad, NPAD);
  zero_bf16_kernel<<<padBlocks, 256, 0, stream>>>(tpad, NPAD);
  stage_w_kernel<<<(3 * 36864 + 255) / 256, 256, 0, stream>>>(w0, wf, w3, w0s, wfs, w3s);

  // conv0: dense 3x3 via WMMA -> x0pad (bf16, padded NHWC)
  conv3x3_wmma_kernel<0><<<1024, 128, 0, stream>>>(xpad, w0s, b0, x0pad, nullptr, nullptr);

  mean_kernel<<<Bb * Cc, 256, 0, stream>>>(x0pad, mbuf);
  eca_kernel<<<(Bb * Cc + 255) / 256, 256, 0, stream>>>(mbuf, w2, b2, abuf);
  dw_kernel<<<NDENSE / 256, 256, 0, stream>>>(x0pad, w1, b1, abuf, dyn);

  // 1x1 dynamic-filter GEMM + per-pixel dynamic conv + LeakyReLU -> tpad
  dynconv_kernel<<<2048, 64, 0, stream>>>(dyn, w3s, b3, x0pad, tpad);

  // convf: dense 3x3 via WMMA + bias + residual -> d_out (fp32 NCHW)
  conv3x3_wmma_kernel<1><<<1024, 128, 0, stream>>>(tpad, wfs, bf, nullptr, out, x);

  (void)in_sizes; (void)n_in; (void)out_size; (void)ws_size; (void)o;
}
